// ConvHex_11742440588008
// MI455X (gfx1250) — matrix operand order repaired and checked by reading, not compiled, not run on hardware
//
#include <hip/hip_runtime.h>

typedef __attribute__((ext_vector_type(16))) _Float16 v16h;
typedef __attribute__((ext_vector_type(2)))  _Float16 v2h;
typedef __attribute__((ext_vector_type(8)))  float    v8f;

#define NB    256      // batch
#define CIN   64
#define COUT  64
#define HH    1855
#define KN    6        // neighbors
#define TAPS  7        // 6 neighbors + center
#define MT    32       // h rows per block = 2 WMMA M-tiles
#define NTILE ((HH + MT - 1) / MT)   // 58
#define NFRAG (TAPS * 2)             // 14 K-chunks of 32
#define OTILES (COUT / 16)           // 4

// Pre-split weight fragments in d_ws:
//   layout [otile(4)][frag(14)][s=hi/lo(2)][lane(32)][16 f16]  = 114,688 bytes
#define WS_V16H_IDX(ot, f, s, lane) ((((ot) * NFRAG + (f)) * 2 + (s)) * 32 + (lane))

// ---------------------------------------------------------------------------
// Prologue: split fp32 weights into f16 hi/lo B-fragments, swizzled for
// v_wmma_f32_16x16x32_f16 (B 32x16: lane = 16*khalf + col, K = khalf*16 + j).
// ---------------------------------------------------------------------------
__global__ __launch_bounds__(256) void prep_weights(
    const float* __restrict__ wC,    // (COUT, CIN)
    const float* __restrict__ wN,    // (COUT, CIN, KN)
    v16h*        __restrict__ ws)
{
    const int t = blockIdx.x * 256 + threadIdx.x;   // 0 .. 4*14*32-1 = 1791
    if (t >= OTILES * NFRAG * 32) return;

    const int lane  = t & 31;
    const int f     = (t >> 5) % NFRAG;
    const int ot    = t / (32 * NFRAG);

    const int col   = lane & 15;
    const int khalf = lane >> 4;
    const int o     = ot * 16 + col;
    const int tap   = f >> 1;
    const int kc    = f & 1;
    const int cbase = kc * 32 + khalf * 16;

    v16h bhi, blo;
    if (tap < KN) {
        const float* wp = wN + ((size_t)o * CIN + cbase) * KN + tap;
        #pragma unroll
        for (int j = 0; j < 16; ++j) {
            float v = wp[j * KN];
            _Float16 h = (_Float16)v;
            bhi[j] = h; blo[j] = (_Float16)(v - (float)h);
        }
    } else {
        const float* wp = wC + o * CIN + cbase;
        #pragma unroll
        for (int j = 0; j < 16; ++j) {
            float v = wp[j];
            _Float16 h = (_Float16)v;
            bhi[j] = h; blo[j] = (_Float16)(v - (float)h);
        }
    }
    ws[WS_V16H_IDX(ot, f, 0, lane)] = bhi;
    ws[WS_V16H_IDX(ot, f, 1, lane)] = blo;
}

// ---------------------------------------------------------------------------
// Main: per block, D(32h x 64o) = A(32h x 448c) * B(448c x 64o), fp32 emulated
// as f16 hi/lo split: A*B ~= Ahi*Bhi + Ahi*Blo + Alo*Bhi  (rel err ~2^-22).
// 4 waves (one 16-wide o-tile each); each B-fragment pair feeds 6 WMMAs
// (two M-tiles) to amortize weight refetch.
// ---------------------------------------------------------------------------
__global__ __launch_bounds__(128) void convhex_wmma_f16x3(
    const float* __restrict__ x,     // (B, CIN, H)
    const int*   __restrict__ nbr,   // (H, KN)
    const v16h*  __restrict__ ws,    // pre-split weight fragments
    const float* __restrict__ bias,  // (COUT)
    float*       __restrict__ out)   // (B, COUT, H)
{
    // A fragments, swizzled for the 16-bit 16x32 A layout: [s][mtile][f][lane][16]
    __shared__ _Float16 shA[2][2][NFRAG][32][16] __attribute__((aligned(32))); // 56KB

    const int tid = threadIdx.x;
    const int b   = blockIdx.x / NTILE;
    const int h0  = (blockIdx.x % NTILE) * MT;

    // ---- stage gathered feature tile (32 x 448) into LDS as f16 hi/lo ----
    // Thread owns one h row (tid&31); neighbor index loaded once per tap.
    // c processed in even pairs -> v_cvt_pk_f16_f32 + single ds_store_b32.
    {
        const int h  = tid & 31;
        const int m  = h >> 4;                   // which M-tile
        const int hl = h & 15;
        const int cs = (tid >> 5) << 1;          // 0,2,4,6
        const int hg = h0 + h;
        const float* xb = x + (size_t)b * CIN * HH;

        #pragma unroll
        for (int tap = 0; tap < TAPS; ++tap) {
            int src = -1;
            if (hg < HH) src = (tap < KN) ? nbr[hg * KN + tap] : hg;

            #pragma unroll
            for (int it = 0; it < 8; ++it) {
                const int c0 = cs + it * 8;      // even; pair (c0, c0+1)
                float v0 = 0.0f, v1 = 0.0f;
                if (src >= 0) {
                    v0 = xb[(size_t)c0 * HH + src];
                    v1 = xb[(size_t)(c0 + 1) * HH + src];
                }
                _Float16 h0f = (_Float16)v0, h1f = (_Float16)v1;
                _Float16 l0f = (_Float16)(v0 - (float)h0f);
                _Float16 l1f = (_Float16)(v1 - (float)h1f);

                // (hl, c) -> (lane, elem): K = half*16 + group*8 + j
                const int kc    = c0 >> 5;
                const int ci    = c0 & 31;
                const int j     = ci & 7;        // even
                const int half  = (ci >> 4) & 1; // pair never straddles
                const int group = (ci >> 3) & 1;
                const int lane  = hl + (half << 4);
                const int e0    = (group << 3) | j;
                const int f     = tap * 2 + kc;

                v2h hp = {h0f, h1f};
                v2h lp = {l0f, l1f};
                *(v2h*)&shA[0][m][f][lane][e0] = hp;
                *(v2h*)&shA[1][m][f][lane][e0] = lp;
            }
        }
    }
    __syncthreads();

    // ---- compute: wave w owns output-channel tile [16w, 16w+16) ----
    const int wave  = tid >> 5;
    const int lane  = tid & 31;
    const int col   = lane & 15;
    const int khalf = lane >> 4;
    const int o     = wave * 16 + col;

    // total_valid = (#valid neighbors of h=0) + 1  (uniform scalar)
    int tv = 1;
    #pragma unroll
    for (int k = 0; k < KN; ++k) tv += (nbr[k] >= 0) ? 1 : 0;
    const float scale = 1.0f / (float)tv;

    v8f acc[2];
    acc[0] = (v8f){0.f, 0.f, 0.f, 0.f, 0.f, 0.f, 0.f, 0.f};
    acc[1] = (v8f){0.f, 0.f, 0.f, 0.f, 0.f, 0.f, 0.f, 0.f};

    #pragma unroll
    for (int f = 0; f < NFRAG; ++f) {
        const v16h bhi = ws[WS_V16H_IDX(wave, f, 0, lane)];
        const v16h blo = ws[WS_V16H_IDX(wave, f, 1, lane)];

        #pragma unroll
        for (int m = 0; m < 2; ++m) {
            const v16h ahi = *(const v16h*)&shA[0][m][f][lane][0];
            const v16h alo = *(const v16h*)&shA[1][m][f][lane][0];
            acc[m] = __builtin_amdgcn_wmma_f32_16x16x32_f16(
                         false, ahi, false, bhi, (short)0, acc[m], false, false);
            acc[m] = __builtin_amdgcn_wmma_f32_16x16x32_f16(
                         false, ahi, false, blo, (short)0, acc[m], false, false);
            acc[m] = __builtin_amdgcn_wmma_f32_16x16x32_f16(
                         false, alo, false, bhi, (short)0, acc[m], false, false);
        }
    }

    // ---- epilogue: lane holds col N=o, rows M = g + 8*khalf -> consecutive h
    const float bo = bias[o];
    #pragma unroll
    for (int m = 0; m < 2; ++m) {
        const int hbase = h0 + m * 16 + 8 * khalf;
        float* op = out + ((size_t)b * COUT + o) * HH + hbase;

        float r[8];
        #pragma unroll
        for (int g = 0; g < 8; ++g) r[g] = acc[m][g] * scale + bo;

        if (hbase + 8 <= HH) {
            *(float4*)(op + 0) = make_float4(r[0], r[1], r[2], r[3]);
            *(float4*)(op + 4) = make_float4(r[4], r[5], r[6], r[7]);
        } else {
            #pragma unroll
            for (int g = 0; g < 8; ++g)
                if (hbase + g < HH) op[g] = r[g];
        }
    }
}

extern "C" void kernel_launch(void* const* d_in, const int* in_sizes, int n_in,
                              void* d_out, int out_size, void* d_ws, size_t ws_size,
                              hipStream_t stream) {
    const float* x    = (const float*)d_in[0];
    const int*   nbr  = (const int*)  d_in[1];
    const float* wC   = (const float*)d_in[2];
    const float* wN   = (const float*)d_in[3];
    const float* bias = (const float*)d_in[4];
    float* out = (float*)d_out;
    v16h* ws   = (v16h*)d_ws;        // needs 114,688 B of scratch

    // 1) split + swizzle weights once (tiny, L2-resident)
    hipLaunchKernelGGL(prep_weights, dim3(7), dim3(256), 0, stream, wC, wN, ws);

    // 2) main GEMM-with-gather: 256 batches x 58 h-tiles of 32 rows
    hipLaunchKernelGGL(convhex_wmma_f16x3, dim3(NB * NTILE), dim3(128), 0, stream,
                       x, nbr, ws, bias, out);
}
